// LinearAttention_1803886265728
// MI455X (gfx1250) — matrix operand order repaired and checked by reading, not compiled, not run on hardware
//
#include <hip/hip_runtime.h>

typedef __attribute__((ext_vector_type(16))) _Float16 v16h;
typedef __attribute__((ext_vector_type(8)))  _Float16 v8h;
typedef __attribute__((ext_vector_type(8)))  float    v8f;

namespace {
constexpr int B_   = 4;
constexpr int N_   = 16384;          // tokens per batch (power of 2)
constexpr int DIM  = 256;
constexpr int H    = 4;
constexpr int DH   = 32;
constexpr int HID  = 128;            // H*DH
constexpr int NMEM = 4;
constexpr int QKV  = 3 * HID;        // 384
constexpr int M    = B_ * N_;        // 65536 tokens
constexpr float SCALE = 0.17677669529663687f; // 32^-0.5
}

union AFrag { v16h v; v8h h[2]; };

__device__ inline float wave_max32(float v) {
#pragma unroll
  for (int o = 16; o > 0; o >>= 1) v = fmaxf(v, __shfl_xor(v, o, 32));
  return v;
}
__device__ inline float wave_sum32(float v) {
#pragma unroll
  for (int o = 16; o > 0; o >>= 1) v += __shfl_xor(v, o, 32);
  return v;
}

// ---------------------------------------------------------------------------
// K1: pre-norm rmsnorm(x, g_pre) -> xn (f16, row-major [m][256]).
// ---------------------------------------------------------------------------
__global__ __launch_bounds__(256) void k_prenorm(const float* __restrict__ x,
                                                 const float* __restrict__ g,
                                                 _Float16* __restrict__ xn) {
  const int w = threadIdx.x >> 5, l = threadIdx.x & 31;
  const size_t m = (size_t)blockIdx.x * 8 + w;
  const float* row = x + m * DIM;
  float v[8];
  float ss = 0.f;
#pragma unroll
  for (int i = 0; i < 8; ++i) { v[i] = row[i * 32 + l]; ss += v[i] * v[i]; }
  ss = wave_sum32(ss);
  const float f = 16.0f / fmaxf(sqrtf(ss), 1e-12f);   // sqrt(256)/||x||
#pragma unroll
  for (int i = 0; i < 8; ++i)
    xn[m * DIM + i * 32 + l] = (_Float16)(v[i] * f * g[i * 32 + l]);
}

// ---------------------------------------------------------------------------
// K1b: transpose+convert w_qkv [256][384] f32 -> wqT [384][256] f16.
// ---------------------------------------------------------------------------
__global__ __launch_bounds__(256) void k_wqkv_T(const float* __restrict__ w,
                                                _Float16* __restrict__ wqT) {
  const int idx = blockIdx.x * 256 + threadIdx.x;  // idx = c*DIM + k
  const int k = idx & (DIM - 1);
  const int c = idx >> 8;
  wqT[idx] = (_Float16)w[(size_t)k * QKV + c];
}

// ---------------------------------------------------------------------------
// K2: QKV GEMM, LDS-free, software-pipelined (double-buffered A/B fragments).
// Wave tile 32(M) x 64(N); block = 8 waves = 256 rows.
// Writes q (f16 [m][hd]), kT (f32 [b*128+hd][n]), vT (f16 [b*128+hd][n]).
// ---------------------------------------------------------------------------
__global__ __launch_bounds__(256) void k_qkv_gemm(const _Float16* __restrict__ xn,
                                                  const _Float16* __restrict__ wqT,
                                                  _Float16* __restrict__ q16,
                                                  float* __restrict__ kT,
                                                  _Float16* __restrict__ vT) {
  const int tid = threadIdx.x, w = tid >> 5, lane = tid & 31;
  const int lr = lane & 15, laneHi = lane >> 4;
  const int M0 = blockIdx.x * 256 + w * 32;   // this wave's 32-row base
  const int n0 = blockIdx.y * 64;
  constexpr int KS = DIM / 32;

  const _Float16* a0p = xn + (size_t)(M0 + lr) * DIM + laneHi * 16;
  const _Float16* a1p = a0p + (size_t)16 * DIM;
  const _Float16* bp  = wqT + (size_t)(n0 + lr) * DIM + laneHi * 16;

  AFrag a0[2], a1[2];
  v16h  bf[2][4];
  auto loadA = [&](int buf, int k0) {
    a0[buf].h[0] = *(const v8h*)(a0p + k0);
    a0[buf].h[1] = *(const v8h*)(a0p + k0 + 8);
    a1[buf].h[0] = *(const v8h*)(a1p + k0);
    a1[buf].h[1] = *(const v8h*)(a1p + k0 + 8);
  };
  auto loadB = [&](int buf, int k0) {
#pragma unroll
    for (int j = 0; j < 4; ++j)
      bf[buf][j] = *(const v16h*)(bp + (size_t)j * 16 * DIM + k0);
  };

  loadA(0, 0);
  loadB(0, 0);
  v8f acc[2][4] = {};
#pragma unroll
  for (int ks = 0; ks < KS; ++ks) {
    const int cur = ks & 1, nxt = cur ^ 1;
    if (ks + 1 < KS) {                         // prefetch next K-step into regs
      loadA(nxt, (ks + 1) * 32);
      loadB(nxt, (ks + 1) * 32);
      __builtin_prefetch(a0p + (ks + 2) * 32, 0, 3);
      __builtin_prefetch(a1p + (ks + 2) * 32, 0, 3);
    }
#pragma unroll
    for (int j = 0; j < 4; ++j) {
      acc[0][j] = __builtin_amdgcn_wmma_f32_16x16x32_f16(false, a0[cur].v, false, bf[cur][j],
                                                         (short)0, acc[0][j], false, false);
      acc[1][j] = __builtin_amdgcn_wmma_f32_16x16x32_f16(false, a1[cur].v, false, bf[cur][j],
                                                         (short)0, acc[1][j], false, false);
    }
  }

  // C layout: row = base + r + 8*laneHi, col = 16j + lr.  gc-range uniform per block.
#pragma unroll
  for (int mi = 0; mi < 2; ++mi) {
#pragma unroll
    for (int j = 0; j < 4; ++j) {
      const int gc = n0 + j * 16 + lr;
#pragma unroll
      for (int r = 0; r < 8; ++r) {
        const size_t m = (size_t)(M0 + mi * 16 + r + laneHi * 8);
        const float val = acc[mi][j][r];
        if (gc < HID) {
          q16[m * HID + gc] = (_Float16)val;
        } else if (gc < 2 * HID) {
          const int hd = gc - HID;
          kT[(size_t)((m >> 14) * HID + hd) * N_ + (m & (N_ - 1))] = val;
        } else {
          const int hd = gc - 2 * HID;
          vT[(size_t)((m >> 14) * HID + hd) * N_ + (m & (N_ - 1))] = (_Float16)val;
        }
      }
    }
  }
}

// ---------------------------------------------------------------------------
// K3: per-(b,h,d) softmax stats over token axis (incl. memory keys).
// ---------------------------------------------------------------------------
__global__ __launch_bounds__(256) void k_kstats(const float* __restrict__ kT,
                                                const float* __restrict__ mem_kv,
                                                float* __restrict__ rowmax,
                                                float* __restrict__ rowsuminv) {
  __shared__ float red[256];
  const int rs = blockIdx.x;               // b*128 + h*32 + d
  const int tid = threadIdx.x;
  const size_t base = (size_t)rs * N_;
  const int hd = rs & 127, h = hd >> 5, d = hd & 31;

  float lm = -3.4e38f;
  for (int i = tid; i < N_; i += 256) lm = fmaxf(lm, kT[base + i]);
  if (tid == 0) {
#pragma unroll
    for (int j = 0; j < NMEM; ++j)
      lm = fmaxf(lm, mem_kv[(h * DH + d) * NMEM + j]);
  }
  red[tid] = lm; __syncthreads();
  for (int s = 128; s > 0; s >>= 1) {
    if (tid < s) red[tid] = fmaxf(red[tid], red[tid + s]);
    __syncthreads();
  }
  const float mx = red[0];
  __syncthreads();

  float lsum = 0.f;
  for (int i = tid; i < N_; i += 256) lsum += expf(kT[base + i] - mx);
  if (tid == 0) {
#pragma unroll
    for (int j = 0; j < NMEM; ++j)
      lsum += expf(mem_kv[(h * DH + d) * NMEM + j] - mx);
  }
  red[tid] = lsum; __syncthreads();
  for (int s = 128; s > 0; s >>= 1) {
    if (tid < s) red[tid] += red[tid + s];
    __syncthreads();
  }
  if (tid == 0) { rowmax[rs] = mx; rowsuminv[rs] = 1.0f / red[0]; }
}

// ---------------------------------------------------------------------------
// K4: context[b,h] = softmax_n(k) @ v^T via WMMA over n.
// Unrolled x2 so exp/cvt VALU of step i+1 co-executes with WMMA of step i.
// ---------------------------------------------------------------------------
__global__ __launch_bounds__(128) void k_ctx(const float* __restrict__ kT,
                                             const _Float16* __restrict__ vT,
                                             const float* __restrict__ rowmax,
                                             const float* __restrict__ rowsuminv,
                                             const float* __restrict__ mem_kv,
                                             float* __restrict__ ctx) {
  const int bh = blockIdx.x;
  const int b = bh / H, h = bh % H;
  const int tid = threadIdx.x, w = tid >> 5, lane = tid & 31;
  const int lr = lane & 15, laneHi = lane >> 4;
  const int dtile = (w & 1) * 16, etile = (w >> 1) * 16;

  const int dg = dtile + lr;                       // A-fragment row for this lane
  const int rs = bh * 32 + dg;                     // == b*128 + h*32 + dg
  const float rmax = rowmax[rs], rinv = rowsuminv[rs];
  const float* kp = kT + (size_t)(b * HID + h * DH + dg) * N_ + laneHi * 16;
  const _Float16* vp = vT + (size_t)(b * HID + h * DH + etile + lr) * N_ + laneHi * 16;

  v8f acc = {};
#pragma unroll 2
  for (int n0 = 0; n0 < N_; n0 += 32) {
    const v8f c0 = *(const v8f*)(kp + n0);
    const v8f c1 = *(const v8f*)(kp + n0 + 8);
    v16h a;
#pragma unroll
    for (int i = 0; i < 8; ++i) {
      a[i]     = (_Float16)(expf(c0[i] - rmax) * rinv);
      a[i + 8] = (_Float16)(expf(c1[i] - rmax) * rinv);
    }
    const v16h bfv = *(const v16h*)(vp + n0);
    acc = __builtin_amdgcn_wmma_f32_16x16x32_f16(false, a, false, bfv,
                                                 (short)0, acc, false, false);
  }

  // Memory-token contribution + store (f32 precision).
#pragma unroll
  for (int r = 0; r < 8; ++r) {
    const int d = dtile + r + laneHi * 8;
    const int e = etile + lr;
    const int rsd = bh * 32 + d;
    const float rm = rowmax[rsd], ri = rowsuminv[rsd];
    float add = 0.f;
#pragma unroll
    for (int j = 0; j < NMEM; ++j) {
      const float mk = mem_kv[(h * DH + d) * NMEM + j];
      const float mv = mem_kv[H * DH * NMEM + (h * DH + e) * NMEM + j];
      add += expf(mk - rm) * ri * mv;
    }
    ctx[bh * (DH * DH) + d * DH + e] = acc[r] + add;
  }
}

// ---------------------------------------------------------------------------
// K5a: q softmax over feature dim (per head, 32 lanes) * SCALE -> qn (f16)
// ---------------------------------------------------------------------------
__global__ __launch_bounds__(256) void k_qsoftmax(const _Float16* __restrict__ q16,
                                                  _Float16* __restrict__ qn16) {
  const int w = threadIdx.x >> 5, l = threadIdx.x & 31;
  const size_t m = (size_t)blockIdx.x * 8 + w;
#pragma unroll
  for (int hh = 0; hh < H; ++hh) {
    const float v = (float)q16[m * HID + hh * DH + l];
    const float mx = wave_max32(v);
    const float e = expf(v - mx);
    const float s = wave_sum32(e);
    qn16[m * HID + hh * DH + l] = (_Float16)(e / s * SCALE);
  }
}

// ---------------------------------------------------------------------------
// K5b: W_comb[b] = BlockDiag(ctx[b]) @ w_out, stored TRANSPOSED:
// wcT[b][c][k] f16 so B-fragments are contiguous per lane.
// ---------------------------------------------------------------------------
__global__ __launch_bounds__(256) void k_wcomb(const float* __restrict__ ctx,
                                               const float* __restrict__ w_out,
                                               _Float16* __restrict__ wcT) {
  const int idx = blockIdx.x * 256 + threadIdx.x;  // ((b*256 + c)*128 + k)
  const int k = idx & 127;
  const int c = (idx >> 7) & 255;
  const int b = idx >> 15;
  const int h = k >> 5, d = k & 31;
  float acc = 0.f;
#pragma unroll
  for (int e = 0; e < DH; ++e)
    acc += ctx[(b * H + h) * (DH * DH) + d * DH + e] * w_out[(h * DH + e) * DIM + c];
  wcT[(size_t)idx] = (_Float16)acc;
}

// ---------------------------------------------------------------------------
// K5c: y = qn @ W_comb[b]  (M x 128 x 256), LDS-free pipelined WMMA.
// y staged in d_out (f32).
// ---------------------------------------------------------------------------
__global__ __launch_bounds__(256) void k_out_gemm(const _Float16* __restrict__ qn,
                                                  const _Float16* __restrict__ wcT,
                                                  float* __restrict__ y) {
  const int tid = threadIdx.x, w = tid >> 5, lane = tid & 31;
  const int lr = lane & 15, laneHi = lane >> 4;
  const int M0 = blockIdx.x * 256 + w * 32;
  const int n0 = blockIdx.y * 64;
  const int b = M0 >> 14;                          // whole wave is one batch
  constexpr int KS = HID / 32;

  const _Float16* a0p = qn + (size_t)(M0 + lr) * HID + laneHi * 16;
  const _Float16* a1p = a0p + (size_t)16 * HID;
  const _Float16* bp  = wcT + (size_t)(b * DIM + n0 + lr) * HID + laneHi * 16;

  AFrag a0[2], a1[2];
  v16h  bf[2][4];
  auto loadA = [&](int buf, int k0) {
    a0[buf].h[0] = *(const v8h*)(a0p + k0);
    a0[buf].h[1] = *(const v8h*)(a0p + k0 + 8);
    a1[buf].h[0] = *(const v8h*)(a1p + k0);
    a1[buf].h[1] = *(const v8h*)(a1p + k0 + 8);
  };
  auto loadB = [&](int buf, int k0) {
#pragma unroll
    for (int j = 0; j < 4; ++j)
      bf[buf][j] = *(const v16h*)(bp + (size_t)j * 16 * HID + k0);
  };

  loadA(0, 0);
  loadB(0, 0);
  v8f acc[2][4] = {};
#pragma unroll
  for (int ks = 0; ks < KS; ++ks) {
    const int cur = ks & 1, nxt = cur ^ 1;
    if (ks + 1 < KS) {
      loadA(nxt, (ks + 1) * 32);
      loadB(nxt, (ks + 1) * 32);
    }
#pragma unroll
    for (int j = 0; j < 4; ++j) {
      acc[0][j] = __builtin_amdgcn_wmma_f32_16x16x32_f16(false, a0[cur].v, false, bf[cur][j],
                                                         (short)0, acc[0][j], false, false);
      acc[1][j] = __builtin_amdgcn_wmma_f32_16x16x32_f16(false, a1[cur].v, false, bf[cur][j],
                                                         (short)0, acc[1][j], false, false);
    }
  }
#pragma unroll
  for (int mi = 0; mi < 2; ++mi) {
#pragma unroll
    for (int j = 0; j < 4; ++j) {
      const int gc = n0 + j * 16 + lr;
#pragma unroll
      for (int r = 0; r < 8; ++r) {
        const size_t m = (size_t)(M0 + mi * 16 + r + laneHi * 8);
        y[m * DIM + gc] = acc[mi][j][r];
      }
    }
  }
}

// ---------------------------------------------------------------------------
// K6: out = rmsnorm(y + b_out, g_out) + x   (in place on d_out, wave/token)
// ---------------------------------------------------------------------------
__global__ __launch_bounds__(256) void k_epilogue(const float* __restrict__ x,
                                                  const float* __restrict__ b_out,
                                                  const float* __restrict__ g_out,
                                                  float* __restrict__ out) {
  const int w = threadIdx.x >> 5, l = threadIdx.x & 31;
  const size_t m = (size_t)blockIdx.x * 8 + w;
  float v[8];
  float ss = 0.f;
#pragma unroll
  for (int i = 0; i < 8; ++i) {
    v[i] = out[m * DIM + i * 32 + l] + b_out[i * 32 + l];
    ss += v[i] * v[i];
  }
  ss = wave_sum32(ss);
  const float f = 16.0f / fmaxf(sqrtf(ss), 1e-12f);
#pragma unroll
  for (int i = 0; i < 8; ++i)
    out[m * DIM + i * 32 + l] = v[i] * f * g_out[i * 32 + l] + x[m * DIM + i * 32 + l];
}

// ---------------------------------------------------------------------------
extern "C" void kernel_launch(void* const* d_in, const int* in_sizes, int n_in,
                              void* d_out, int out_size, void* d_ws, size_t ws_size,
                              hipStream_t stream) {
  (void)in_sizes; (void)n_in; (void)out_size; (void)ws_size;
  const float* x      = (const float*)d_in[0];
  /* d_in[1] = t, unused */
  const float* w_qkv  = (const float*)d_in[2];
  const float* mem_kv = (const float*)d_in[3];
  const float* w_out  = (const float*)d_in[4];
  const float* b_out  = (const float*)d_in[5];
  const float* g_out  = (const float*)d_in[6];
  const float* g_pre  = (const float*)d_in[7];
  float* out = (float*)d_out;

  char* p = (char*)d_ws;
  auto take = [&](size_t bytes) { char* r = p; p += (bytes + 255) & ~(size_t)255; return r; };
  _Float16* xn   = (_Float16*)take((size_t)M * DIM * 2);
  _Float16* wqT  = (_Float16*)take((size_t)DIM * QKV * 2);
  _Float16* q16  = (_Float16*)take((size_t)M * HID * 2);
  float*    kT   = (float*)   take((size_t)B_ * HID * N_ * 4);
  _Float16* vT   = (_Float16*)take((size_t)B_ * HID * N_ * 2);
  float*    rmax = (float*)   take((size_t)B_ * HID * 4);
  float*    rsum = (float*)   take((size_t)B_ * HID * 4);
  float*    ctx  = (float*)   take((size_t)B_ * H * DH * DH * 4);
  _Float16* qn16 = (_Float16*)take((size_t)M * HID * 2);
  _Float16* wcT  = (_Float16*)take((size_t)B_ * DIM * HID * 2);

  k_prenorm  <<<M / 8, 256, 0, stream>>>(x, g_pre, xn);
  k_wqkv_T   <<<(QKV * DIM) / 256, 256, 0, stream>>>(w_qkv, wqT);
  k_qkv_gemm <<<dim3(M / 256, QKV / 64), 256, 0, stream>>>(xn, wqT, q16, kT, vT);
  k_kstats   <<<B_ * HID, 256, 0, stream>>>(kT, mem_kv, rmax, rsum);
  k_ctx      <<<B_ * H, 128, 0, stream>>>(kT, vT, rmax, rsum, mem_kv, ctx);
  k_qsoftmax <<<M / 8, 256, 0, stream>>>(q16, qn16);
  k_wcomb    <<<(B_ * DIM * HID) / 256, 256, 0, stream>>>(ctx, w_out, wcT);
  k_out_gemm <<<dim3(M / 256, DIM / 64), 256, 0, stream>>>(qn16, wcT, out);
  k_epilogue <<<M / 8, 256, 0, stream>>>(x, b_out, g_out, out);
}
